// Retinal_NET_27788438405577
// MI455X (gfx1250) — matrix operand order repaired and checked by reading, not compiled, not run on hardware
//
#include <hip/hip_runtime.h>

typedef __attribute__((ext_vector_type(16))) _Float16 v16h;
typedef __attribute__((ext_vector_type(8)))  _Float16 v8h;
typedef __attribute__((ext_vector_type(8)))  float    v8f;

#define NB 16
#define TT 100
#define TP 128     // padded time (K dim), zero-filled 100..127
#define S2 2500    // 50*50
#define S2P 2512   // padded spatial = 157*16
#define W50 50

// Workspace layout (bytes, 256-aligned)
#define S1LIN_OFF 0u          //  8,000,000  f16 [b][t][s]      (stage-1 out)
#define S1T_OFF   8000000u    // 10,289,152  f16 [b][s][128]    (transposed, padded)
#define ATILE_OFF 18290176u   //     28,672  f16 A tiles [7][4][32][16]
#define FIRST_OFF 18319104u   // 16,000,000  f32 [b][t][s]      (stage-2 out)
#define FLAT_OFF  34319360u   //    640,000  f32 [b][t][100]
#define AMA_OFF   34959616u   //    313,600  f32
#define GANG_OFF  35273472u   //    313,600  f32

// -------------------------------------------------------------------------
// K1: 5x5 stride-5 spatial conv on (16,1,100,250,250) -> f16 [b][t][2500].
// The bandwidth-bound stage: one streaming pass over the 400 MB input.
// -------------------------------------------------------------------------
__global__ void space_conv_kernel(const float* __restrict__ x,
                                  const float* __restrict__ w_space,
                                  const float* __restrict__ b_space,
                                  _Float16* __restrict__ s1h) {
  unsigned idx = blockIdx.x * blockDim.x + threadIdx.x;
  const unsigned total = NB * TT * S2;
  if (idx >= total) return;
  unsigned xo = idx % W50;
  unsigned yo = (idx / W50) % W50;
  unsigned bt = idx / S2;                       // b*100 + t
  const float* base = x + (size_t)bt * 62500u + (size_t)yo * 5u * 250u + xo * 5u;
  if (bt + 1u < (unsigned)(NB * TT))
    __builtin_prefetch(base + 62500, 0, 1);     // next temporal slice
  float acc = b_space[0];
#pragma unroll
  for (int i = 0; i < 5; ++i) {
#pragma unroll
    for (int j = 0; j < 5; ++j)
      acc += base[i * 250 + j] * w_space[i * 5 + j];
  }
  s1h[idx] = (_Float16)acc;
}

// -------------------------------------------------------------------------
// K2a: materialize Toeplitz A tiles in the exact WMMA per-lane layout.
// a_tiles[tile][kc][lane][16 halves]; zero outside the 51-tap band.
// 16-bit A 16x32 layout used here (contiguous k): lanes 0-15 K={0..15},
// lanes 16-31 K={16..31}; VGPR j holds K pair 16*hi+2j, 16*hi+2j+1.
// -------------------------------------------------------------------------
__global__ void build_a_tiles_kernel(const float* __restrict__ w_temporal,
                                     _Float16* __restrict__ a_tiles) {
  int idx = blockIdx.x * blockDim.x + threadIdx.x;   // 7*4*32 = 896
  if (idx >= 7 * 4 * 32) return;
  int lane = idx & 31;
  int kc   = (idx >> 5) & 3;
  int tile = idx >> 7;
  bool hi  = lane >= 16;
  int t    = tile * 16 + (lane & 15);
  int kb   = kc * 32 + (hi ? 16 : 0);
  _Float16* dst = a_tiles + (size_t)idx * 16;
#pragma unroll
  for (int j = 0; j < 8; ++j) {
    int kk = kb + 2 * j;
#pragma unroll
    for (int h = 0; h < 2; ++h) {
      int k = kk + h - t + 50;                       // tap index
      dst[2 * j + h] = (k >= 0 && k <= 50) ? (_Float16)w_temporal[k]
                                           : (_Float16)0.0f;
    }
  }
}

// -------------------------------------------------------------------------
// K2b: LDS-tiled transpose [b][t][2500] -> [b][s][128] (zero-padded).
// Makes every B-operand fetch in K2c one contiguous 32-byte (16-half) load.
// -------------------------------------------------------------------------
__global__ __launch_bounds__(256)
void transpose_kernel(const _Float16* __restrict__ in,
                      _Float16* __restrict__ out) {
  __shared__ _Float16 tile[TP][16 + 2];
  const int b  = blockIdx.y;
  const int s0 = blockIdx.x * 16;                    // 157 tiles -> 2512
  const int tid = threadIdx.x;
  const int sl = tid & 15;
  const int tl = tid >> 4;                           // 0..15
  const _Float16* inb = in + (size_t)b * TT * S2;
#pragma unroll
  for (int k = 0; k < 8; ++k) {
    int t = tl + 16 * k;                             // 0..127
    int s = s0 + sl;
    tile[t][sl] = (t < TT && s < S2) ? inb[(size_t)t * S2 + s] : (_Float16)0.0f;
  }
  __syncthreads();
  const int sr = tid >> 4;                           // output row in tile
  const int tc = (tid & 15) * 8;                     // 8 consecutive t
  v8h v;
#pragma unroll
  for (int q = 0; q < 8; ++q) v[q] = tile[tc + q][sr];
  _Float16* outb = out + (size_t)b * S2P * TP;
  *(v8h*)(outb + (size_t)(s0 + sr) * TP + tc) = v;   // 16B-aligned, coalesced
}

// -------------------------------------------------------------------------
// K2c: 51-tap causal temporal conv as banded-Toeplitz GEMM via WMMA.
// First[t,s] = relu( sum_{t'} Wtoep[t,t'] * S1[t',s] + b_temporal ).
// One wave per 16(t) x 16(s) tile; K fully unrolled over 4 chunks of 32
// (out-of-band chunks have all-zero A). Inner body: 2 vector loads + WMMA.
// -------------------------------------------------------------------------
__global__ __launch_bounds__(32)
void temporal_wmma_kernel(const _Float16* __restrict__ s1t,
                          const _Float16* __restrict__ a_tiles,
                          const float* __restrict__ b_temporal,
                          float* __restrict__ first) {
  const int lane = threadIdx.x;                      // wave32
  const int s0 = blockIdx.x * 16;                    // 0..2496
  const int tileM = blockIdx.y;                      // 0..6
  const int b  = blockIdx.z;

  const bool hi = lane >= 16;
  const int n = s0 + (lane & 15);                    // B/D column (spatial)
  const _Float16* brow = s1t + (size_t)b * S2P * TP + (size_t)n * TP + (hi ? 16 : 0);
  const v16h* atp = (const v16h*)a_tiles + (size_t)(tileM * 4) * 32 + lane;

  v8f acc = {};
#pragma unroll
  for (int kc = 0; kc < 4; ++kc) {
    v16h a  = atp[(size_t)kc * 32];                  // Toeplitz weights
    v16h bm = *(const v16h*)(brow + kc * 32);        // 16 contiguous K halves
    acc = __builtin_amdgcn_wmma_f32_16x16x32_f16(
        false, a, false, bm, (short)0, acc, false, false);
  }

  // D layout: VGPR r -> M = r + (hi ? 8 : 0), N = lane&15.  Bias + ReLU.
  const float btc = b_temporal[0];
  float* outb = first + (size_t)b * TT * S2;
  const int t0 = tileM * 16 + (hi ? 8 : 0);
  if (n < S2) {
#pragma unroll
    for (int r = 0; r < 8; ++r) {
      int tt = t0 + r;
      if (tt < TT) {
        float v = acc[r] + btc;
        outb[(size_t)tt * S2 + n] = v > 0.0f ? v : 0.0f;
      }
    }
  }
}

// -------------------------------------------------------------------------
// K3: 5x5 stride-5 conv on (16,100,50,50) -> flat (16,100,100)
// -------------------------------------------------------------------------
__global__ void last_conv_kernel(const float* __restrict__ first,
                                 const float* __restrict__ w_last,
                                 const float* __restrict__ b_last,
                                 float* __restrict__ flat) {
  unsigned idx = blockIdx.x * blockDim.x + threadIdx.x;
  const unsigned total = NB * TT * 100;
  if (idx >= total) return;
  unsigned p = idx % 100;                            // r*10 + c
  unsigned bt = idx / 100;
  unsigned c = p % 10, r = p / 10;
  const float* base = first + (size_t)bt * S2 + (size_t)r * 5u * W50 + c * 5u;
  float acc = b_last[0];
#pragma unroll
  for (int i = 0; i < 5; ++i) {
#pragma unroll
    for (int j = 0; j < 5; ++j)
      acc += base[i * W50 + j] * w_last[i * 5 + j];
  }
  flat[idx] = acc;
}

// -------------------------------------------------------------------------
// K4: amacrine / ganglion cell gathers (1x1 convs, stride-2 / roll-25)
// -------------------------------------------------------------------------
__global__ void branch_pre_kernel(const float* __restrict__ flat,
                                  const float* __restrict__ wa, const float* __restrict__ ba,
                                  const float* __restrict__ wg, const float* __restrict__ bg,
                                  float* __restrict__ ama_pre,
                                  float* __restrict__ gang_pre) {
  unsigned idx = blockIdx.x * blockDim.x + threadIdx.x;
  const unsigned total = NB * TT * 49;
  if (idx >= total) return;
  unsigned i = idx % 49;
  unsigned bt = idx / 49;
  const float* f = flat + (size_t)bt * 100u;
  ama_pre[idx]  = f[2u * i]                * wa[0] + ba[0];
  gang_pre[idx] = f[(2u * i + 25u) % 100u] * wg[0] + bg[0];
}

// -------------------------------------------------------------------------
// K5: only t = 99 is consumed; 51-tap convs over pre[49..99], combine,
// 49-wide weighted reduction per batch.
// -------------------------------------------------------------------------
__global__ void readout_kernel(const float* __restrict__ ama_pre,
                               const float* __restrict__ gang_pre,
                               const float* __restrict__ w_amak, const float* __restrict__ b_amak,
                               const float* __restrict__ w_amaa, const float* __restrict__ b_amaa,
                               const float* __restrict__ w_gank, const float* __restrict__ b_gank,
                               const float* __restrict__ w_col,  const float* __restrict__ b_col,
                               float* __restrict__ out) {
  __shared__ float red[64];
  const int b = blockIdx.x;
  const int i = threadIdx.x;                         // 64 threads
  float val = 0.0f;
  if (i < 49) {
    const float* ap = ama_pre  + (size_t)b * TT * 49 + i;
    const float* gp = gang_pre + (size_t)b * TT * 49 + i;
    float as = b_amak[0], gs = b_gank[0];
    for (int k = 0; k < 51; ++k) {
      int tp = 49 + k;
      as += ap[(size_t)tp * 49] * w_amak[k];
      gs += gp[(size_t)tp * 49] * w_gank[k];
    }
    float ama = w_amaa[0] * (as > 0.0f ? as : 0.0f) + b_amaa[0];
    float g = gs - fabsf(ama);
    val = (g > 0.0f ? g : 0.0f) * w_col[i];
  }
  red[i] = val;
  __syncthreads();
  for (int off = 32; off > 0; off >>= 1) {
    if (i < off) red[i] += red[i + off];
    __syncthreads();
  }
  if (i == 0) out[b] = red[0] + b_col[0];
}

// -------------------------------------------------------------------------
extern "C" void kernel_launch(void* const* d_in, const int* in_sizes, int n_in,
                              void* d_out, int out_size, void* d_ws, size_t ws_size,
                              hipStream_t stream) {
  const float* x          = (const float*)d_in[0];
  const float* w_space    = (const float*)d_in[1];
  const float* b_space    = (const float*)d_in[2];
  const float* w_temporal = (const float*)d_in[3];
  const float* b_temporal = (const float*)d_in[4];
  const float* w_last     = (const float*)d_in[5];
  const float* b_last     = (const float*)d_in[6];
  const float* w_amac     = (const float*)d_in[7];
  const float* b_amac     = (const float*)d_in[8];
  const float* w_ganc     = (const float*)d_in[9];
  const float* b_ganc     = (const float*)d_in[10];
  const float* w_amak     = (const float*)d_in[11];
  const float* b_amak     = (const float*)d_in[12];
  const float* w_amaa     = (const float*)d_in[13];
  const float* b_amaa     = (const float*)d_in[14];
  const float* w_gank     = (const float*)d_in[15];
  const float* b_gank     = (const float*)d_in[16];
  const float* w_col      = (const float*)d_in[17];
  const float* b_col      = (const float*)d_in[18];

  char* ws = (char*)d_ws;
  _Float16* s1h_lin = (_Float16*)(ws + S1LIN_OFF);
  _Float16* s1t     = (_Float16*)(ws + S1T_OFF);
  _Float16* a_tiles = (_Float16*)(ws + ATILE_OFF);
  float* first      = (float*)(ws + FIRST_OFF);
  float* flat       = (float*)(ws + FLAT_OFF);
  float* ama_pre    = (float*)(ws + AMA_OFF);
  float* gang_pre   = (float*)(ws + GANG_OFF);

  {
    unsigned total = NB * TT * S2;                   // 4,000,000
    space_conv_kernel<<<(total + 255) / 256, 256, 0, stream>>>(
        x, w_space, b_space, s1h_lin);
  }
  build_a_tiles_kernel<<<4, 256, 0, stream>>>(w_temporal, a_tiles);
  {
    dim3 grid(S2P / 16, NB);                         // 157 x 16
    transpose_kernel<<<grid, 256, 0, stream>>>(s1h_lin, s1t);
  }
  {
    dim3 grid(S2P / 16, 7, NB);                      // s-tiles, t-tiles, batch
    temporal_wmma_kernel<<<grid, 32, 0, stream>>>(
        s1t, a_tiles, b_temporal, first);
  }
  {
    unsigned total = NB * TT * 100;                  // 160,000
    last_conv_kernel<<<(total + 255) / 256, 256, 0, stream>>>(
        first, w_last, b_last, flat);
  }
  {
    unsigned total = NB * TT * 49;                   // 78,400
    branch_pre_kernel<<<(total + 255) / 256, 256, 0, stream>>>(
        flat, w_amac, b_amac, w_ganc, b_ganc, ama_pre, gang_pre);
  }
  readout_kernel<<<NB, 64, 0, stream>>>(
      ama_pre, gang_pre, w_amak, b_amak, w_amaa, b_amaa,
      w_gank, b_gank, w_col, b_col, (float*)d_out);
}
